// FluxDoubleStreamBlock_53884659696161
// MI455X (gfx1250) — matrix operand order repaired and checked by reading, not compiled, not run on hardware
//
#include <hip/hip_runtime.h>

// ---------------------------------------------------------------------------
// FluxDoubleStreamBlock forward for MI455X (gfx1250, wave32, WMMA).
// GEMMs: fp32 global -> packed bf16 (v_perm_b32/v_cvt_pk) -> LDS -> v_wmma.
// Attention: bf16 q/k/vT buffers -> global_load_async_to_lds_b128 (ASYNCcnt).
// ---------------------------------------------------------------------------

#define Hh   24
#define HDd  128
#define DM   3072
#define MLPd 12288
#define LIi  1024
#define LTt  512
#define LLl  1536

typedef __attribute__((ext_vector_type(16))) __bf16        v16bf;
typedef __attribute__((ext_vector_type(8)))  float         v8f;
typedef __attribute__((ext_vector_type(4)))  unsigned int  u32x4;

union FragAB { v16bf v; u32x4 q[2]; };

__device__ __forceinline__ unsigned short f2bf(float f) {
  union { float f; unsigned u; } x; x.f = f;
  return (unsigned short)((x.u + 0x8000u) >> 16);   // round-half-up
}

// Packed fp32x2 -> bf16x2: prefer v_cvt_pk_bf16_f32, else add+v_perm_b32 (3 VALU).
#if defined(__AMDGCN__) && __has_builtin(__builtin_amdgcn_cvt_pk_bf16_f32)
__device__ __forceinline__ unsigned pk2bf(float a, float b) {
  auto r = __builtin_amdgcn_cvt_pk_bf16_f32(a, b);
  union { decltype(r) v; unsigned u; } x; x.v = r;
  return x.u;
}
#elif defined(__AMDGCN__) && __has_builtin(__builtin_amdgcn_perm)
__device__ __forceinline__ unsigned pk2bf(float a, float b) {
  union { float f; unsigned u; } x, y; x.f = a; y.f = b;
  // result bytes [1:0] = (a+0x8000)[3:2] (S1 sel 2,3), [3:2] = (b+0x8000)[3:2] (S0 sel 6,7)
  return __builtin_amdgcn_perm(y.u + 0x8000u, x.u + 0x8000u, 0x07060302u);
}
#else
__device__ __forceinline__ unsigned pk2bf(float a, float b) {
  return (unsigned)f2bf(a) | ((unsigned)f2bf(b) << 16);
}
#endif

__device__ __forceinline__ v8f v8f_zero() {
  v8f z = {0.f,0.f,0.f,0.f,0.f,0.f,0.f,0.f};
  return z;
}

__device__ __forceinline__ v8f wmma_bf16(v16bf a, v16bf b, v8f c) {
  // D = A(16x32 bf16) * B(32x16 bf16) + C(16x16 f32)
  return __builtin_amdgcn_wmma_f32_16x16x32_bf16(false, a, false, b,
                                                 (short)0, c, false, false);
}

// A-fragment (16x32 bf16). LDS tile row-major [m][ld].
__device__ __forceinline__ v16bf load_frag_a(const unsigned short* base, int ld, int lane) {
  int m  = lane & 15;
  int k0 = (lane >> 4) * 8;
  FragAB u;
  u.q[0] = *(const u32x4*)(base + m * ld + k0);
  u.q[1] = *(const u32x4*)(base + m * ld + k0 + 16);
  return u.v;
}

// B-fragment (32x16 bf16). LDS tile stored N-major [n][ld].
__device__ __forceinline__ v16bf load_frag_b(const unsigned short* base, int ld, int lane) {
  int n  = lane & 15;
  int k0 = (lane >> 4) * 8;
  FragAB u;
  u.q[0] = *(const u32x4*)(base + n * ld + k0);
  u.q[1] = *(const u32x4*)(base + n * ld + k0 + 16);
  return u.v;
}

// CDNA5 async global->LDS copy (16B/lane), tracked by ASYNCcnt.
// LDS flat addresses are {aperture, offset[31:0]} -> low 32 bits = DS offset.
__device__ __forceinline__ unsigned lds_addr32(const void* p) {
  return (unsigned)(unsigned long long)p;
}
__device__ __forceinline__ void async_copy_b128(unsigned lds_off, const void* gsrc) {
  asm volatile("global_load_async_to_lds_b128 %0, %1, off"
               : : "v"(lds_off), "v"(gsrc) : "memory");
}
__device__ __forceinline__ void wait_async0() {
  asm volatile("s_wait_asynccnt 0x0" : : : "memory");
}

__device__ __forceinline__ float gelu_tanh(float x) {
  float x3 = x * x * x;
  return 0.5f * x * (1.0f + tanhf(0.7978845608028654f * (x + 0.044715f * x3)));
}

// ---------------------------------------------------------------------------
// Tiled WMMA GEMM: C[M,N] = epilogue(A[M,K] @ B[K,N] + bias)
// BM=BN=128, BK=32, 256 threads = 8 waves (2x4), wave tile 64x32 (4x2 frags).
// mode 0: bias; mode 1: gelu(bias); mode 2: res + gate[n]*(acc+bias)
// ---------------------------------------------------------------------------
__global__ void __launch_bounds__(256) wmma_gemm(
    const float* __restrict__ A, int lda,
    const float* __restrict__ B, int ldb,
    const float* __restrict__ bias,
    float* __restrict__ C, int ldc,
    const float* __restrict__ res,
    const float* __restrict__ gate,
    int K, int mode)
{
  __shared__ unsigned short As[128 * 32] __attribute__((aligned(16)));  // [m][k]
  __shared__ unsigned short Bs[128 * 32] __attribute__((aligned(16)));  // [n][k]

  const int bn   = blockIdx.x * 128;
  const int bm   = blockIdx.y * 128;
  const int tid  = threadIdx.x;
  const int lane = tid & 31;
  const int wave = tid >> 5;
  const int wm   = (wave & 1) * 64;
  const int wn   = (wave >> 1) * 32;

  v8f acc[4][2];
#pragma unroll
  for (int i = 0; i < 4; ++i)
#pragma unroll
    for (int j = 0; j < 2; ++j) acc[i][j] = v8f_zero();

  const int ar  = tid >> 1;         // A stage: row 0..127
  const int ac  = (tid & 1) * 16;   // A stage: k half
  const int bn2 = tid & 127;        // B stage: n column
  const int bkq = (tid >> 7) * 16;  // B stage: k half

  for (int k0 = 0; k0 < K; k0 += 32) {
    {  // stage A 128x32 fp32 -> bf16 LDS, row-major [m][k], packed stores
      const float* src = A + (size_t)(bm + ar) * lda + k0 + ac;
      unsigned* dst = (unsigned*)(As + ar * 32 + ac);
#pragma unroll
      for (int j = 0; j < 16; j += 4) {
        float4 f = *(const float4*)(src + j);
        dst[(j >> 1) + 0] = pk2bf(f.x, f.y);
        dst[(j >> 1) + 1] = pk2bf(f.z, f.w);
      }
      if (k0 + 32 < K) __builtin_prefetch(src + 32, 0, 1);   // global_prefetch_b8
    }
    {  // stage B 32x128 fp32 -> bf16 LDS, transposed [n][k], packed stores
      const float* src = B + (size_t)(k0 + bkq) * ldb + bn + bn2;
      unsigned* dst = (unsigned*)(Bs + bn2 * 32 + bkq);
#pragma unroll
      for (int i = 0; i < 16; i += 2) {
        float f0 = src[(size_t)i * ldb];
        float f1 = src[(size_t)(i + 1) * ldb];
        dst[i >> 1] = pk2bf(f0, f1);
      }
      if (k0 + 32 < K) __builtin_prefetch(src + (size_t)32 * ldb, 0, 1);
    }
    __syncthreads();

    v16bf af[4], bfr[2];
#pragma unroll
    for (int i = 0; i < 4; ++i) af[i]  = load_frag_a(As + (wm + i * 16) * 32, 32, lane);
#pragma unroll
    for (int j = 0; j < 2; ++j) bfr[j] = load_frag_b(Bs + (wn + j * 16) * 32, 32, lane);
#pragma unroll
    for (int i = 0; i < 4; ++i)
#pragma unroll
      for (int j = 0; j < 2; ++j)
        acc[i][j] = wmma_bf16(af[i], bfr[j], acc[i][j]);
    __syncthreads();
  }

  // epilogue: C-layout lane=(N), vgpr v => row v + 8*(lane>>4)
  const int nlo = lane & 15;
  const int hi  = lane >> 4;
#pragma unroll
  for (int i = 0; i < 4; ++i)
#pragma unroll
    for (int j = 0; j < 2; ++j) {
      int n = bn + wn + j * 16 + nlo;
      float bv = bias ? bias[n] : 0.0f;
      float gv = (mode == 2) ? gate[n] : 0.0f;
#pragma unroll
      for (int v = 0; v < 8; ++v) {
        int m = bm + wm + i * 16 + v + 8 * hi;
        float x = acc[i][j][v] + bv;
        float o;
        if (mode == 1)      o = gelu_tanh(x);
        else if (mode == 2) o = res[(size_t)m * ldc + n] + gv * x;
        else                o = x;
        C[(size_t)m * ldc + n] = o;
      }
    }
}

// ---------------------------------------------------------------------------
// silu(vec) + the two modulation GEMVs (1x3072 @ 3072x18432).
// ---------------------------------------------------------------------------
__global__ void __launch_bounds__(256) mod_gemv_kernel(
    const float* __restrict__ vec,
    const float* __restrict__ wi, const float* __restrict__ bi,
    const float* __restrict__ wt, const float* __restrict__ bt,
    float* __restrict__ modv)
{
  __shared__ float sv[DM];
  const int tid = threadIdx.x;
  for (int i = tid; i < DM; i += 256) {
    float a = vec[i];
    sv[i] = a / (1.0f + __expf(-a));
  }
  __syncthreads();
  const int o0 = blockIdx.x * 64;
  const float* w; const float* b; int oc;
  if (o0 < 6 * DM) { w = wi; b = bi; oc = o0; }
  else             { w = wt; b = bt; oc = o0 - 6 * DM; }
  const int olane = tid & 63;
  const int krow  = tid >> 6;
  float acc = 0.0f;
  for (int k = krow; k < DM; k += 4)
    acc += sv[k] * w[(size_t)k * (6 * DM) + oc + olane];
  __shared__ float part[4][64];
  part[krow][olane] = acc;
  __syncthreads();
  if (tid < 64)
    modv[o0 + tid] = part[0][tid] + part[1][tid] + part[2][tid] + part[3][tid] + b[oc + tid];
}

// ---------------------------------------------------------------------------
// LayerNorm (eps 1e-6, no affine) then * (1+scale[n]) + shift[n]. One row/block.
// ---------------------------------------------------------------------------
__global__ void __launch_bounds__(256) ln_mod_kernel(
    const float* __restrict__ x,
    const float* __restrict__ shift,
    const float* __restrict__ scl,
    float* __restrict__ out)
{
  const int row = blockIdx.x;
  const int tid = threadIdx.x;
  const float* xr = x + (size_t)row * DM;
  float s = 0.0f, ss = 0.0f;
  for (int i = tid; i < DM; i += 256) { float v = xr[i]; s += v; ss += v * v; }
  __shared__ float rs[256], rss[256];
  rs[tid] = s; rss[tid] = ss;
  __syncthreads();
  for (int st = 128; st > 0; st >>= 1) {
    if (tid < st) { rs[tid] += rs[tid + st]; rss[tid] += rss[tid + st]; }
    __syncthreads();
  }
  float mean = rs[0] * (1.0f / DM);
  float var  = rss[0] * (1.0f / DM) - mean * mean;
  float inv  = rsqrtf(var + 1e-6f);
  float* orow = out + (size_t)row * DM;
  for (int i = tid; i < DM; i += 256)
    orow[i] = (xr[i] - mean) * inv * (1.0f + scl[i]) + shift[i];
}

// ---------------------------------------------------------------------------
// Per (l,h): RMS-norm q/k (f32 machine eps) + RoPE -> bf16 q/k buffers,
// v -> bf16 pre-transposed [h*128+d][1536] buffer (async-copy friendly).
// ---------------------------------------------------------------------------
__global__ void __launch_bounds__(128) qkv_post_kernel(
    const float* __restrict__ qkv_img,
    const float* __restrict__ qkv_txt,
    const float* __restrict__ nqw,  const float* __restrict__ nkw,
    const float* __restrict__ naqw, const float* __restrict__ nakw,
    const float* __restrict__ pe,
    unsigned short* __restrict__ qb,
    unsigned short* __restrict__ kb,
    unsigned short* __restrict__ vbT)
{
  const int l = blockIdx.x;
  const int h = blockIdx.y;
  const int d = threadIdx.x;
  const float* src; const float* qw; const float* kw;
  if (l < LTt) { src = qkv_txt + (size_t)l * (3 * DM);          qw = naqw; kw = nakw; }
  else         { src = qkv_img + (size_t)(l - LTt) * (3 * DM);  qw = nqw;  kw = nkw;  }

  __shared__ float red[128];
  __shared__ float tmp[128];
  const float eps = 1.1920929e-07f;
  const size_t obase = (size_t)l * DM + h * HDd;

  for (int which = 0; which < 2; ++which) {
    const float* w = which ? kw : qw;
    float x = src[which * DM + h * HDd + d];
    red[d] = x * x;
    __syncthreads();
    for (int s = 64; s > 0; s >>= 1) {
      if (d < s) red[d] += red[d + s];
      __syncthreads();
    }
    float rms = rsqrtf(red[0] * (1.0f / 128.0f) + eps);
    tmp[d] = x * rms * w[d];
    __syncthreads();
    if (d < 64) {
      float x0 = tmp[2 * d], x1 = tmp[2 * d + 1];
      const float* f = pe + ((size_t)l * 64 + d) * 4;  // [l][i][j][c]
      float o0 = f[0] * x0 + f[1] * x1;
      float o1 = f[2] * x0 + f[3] * x1;
      unsigned* dst = (unsigned*)((which ? kb : qb) + obase + 2 * d);
      *dst = pk2bf(o0, o1);
    }
    __syncthreads();
  }
  vbT[(size_t)(h * HDd + d) * LLl + l] = f2bf(src[2 * DM + h * HDd + d]);
}

// ---------------------------------------------------------------------------
// Flash attention, one (64-row q-tile, head) per block. 128 threads = 4 waves,
// each wave owns 16 q-rows. Tiles staged with GLOBAL_LOAD_ASYNC_TO_LDS_B128
// (raw bf16, V already transposed), waited with s_wait_asynccnt.
// ---------------------------------------------------------------------------
__global__ void __launch_bounds__(128) flash_attn_kernel(
    const unsigned short* __restrict__ qb,
    const unsigned short* __restrict__ kb,
    const unsigned short* __restrict__ vbT,
    float* __restrict__ attn)
{
  __shared__ unsigned short Qs[64 * 128]  __attribute__((aligned(16)));  // [q][d]
  __shared__ unsigned short Ks[64 * 128]  __attribute__((aligned(16)));  // [key][d]
  __shared__ unsigned short Vt[128 * 64]  __attribute__((aligned(16)));  // [d][key]
  __shared__ unsigned short Ps[4][16 * 64] __attribute__((aligned(16))); // per-wave P

  const int qt   = blockIdx.x;
  const int h    = blockIdx.y;
  const int tid  = threadIdx.x;
  const int lane = tid & 31;
  const int wave = tid >> 5;
  const int nlo  = lane & 15;
  const int hi   = lane >> 4;

  const int r  = tid >> 1;         // staging row 0..63
  const int dh = (tid & 1) * 64;   // staging d half (64 bf16 = 128B per thread)

  {  // stage Q tile asynchronously (raw bf16 copy)
    const unsigned short* src = qb + (size_t)(qt * 64 + r) * DM + h * HDd + dh;
    unsigned dst = lds_addr32(Qs + r * 128 + dh);
#pragma unroll
    for (int j = 0; j < 8; ++j)
      async_copy_b128(dst + j * 16, src + j * 8);
  }

  v8f O[8];
#pragma unroll
  for (int i = 0; i < 8; ++i) O[i] = v8f_zero();
  float mrow[8], lsum[8];
#pragma unroll
  for (int v = 0; v < 8; ++v) { mrow[v] = -3.0e38f; lsum[v] = 0.0f; }

  const float scale = 0.08838834764831845f;  // 1/sqrt(128)

  for (int kt = 0; kt < 24; ++kt) {
    __syncthreads();   // previous tile fully consumed by all waves
    {  // stage K tile [key][d] asynchronously
      const unsigned short* src = kb + (size_t)(kt * 64 + r) * DM + h * HDd + dh;
      unsigned dst = lds_addr32(Ks + r * 128 + dh);
#pragma unroll
      for (int j = 0; j < 8; ++j)
        async_copy_b128(dst + j * 16, src + j * 8);
    }
    {  // stage V tile [d][key] asynchronously (vbT already transposed)
      const unsigned short* src = vbT + (size_t)(h * HDd + tid) * LLl + kt * 64;
      unsigned dst = lds_addr32(Vt + tid * 64);
#pragma unroll
      for (int j = 0; j < 8; ++j)
        async_copy_b128(dst + j * 16, src + j * 8);
    }
    wait_async0();     // this wave's async copies complete (covers Q on kt==0)
    __syncthreads();   // all waves' tiles visible

    // S(16x64) = Q(16x128) @ K^T, 4 d-chunks x 4 key-frags
    v8f S[4];
#pragma unroll
    for (int nf = 0; nf < 4; ++nf) S[nf] = v8f_zero();
#pragma unroll
    for (int dc = 0; dc < 4; ++dc) {
      v16bf aq = load_frag_a(Qs + (wave * 16) * 128 + dc * 32, 128, lane);
#pragma unroll
      for (int nf = 0; nf < 4; ++nf) {
        v16bf bk = load_frag_b(Ks + (nf * 16) * 128 + dc * 32, 128, lane);
        S[nf] = wmma_bf16(aq, bk, S[nf]);
      }
    }

    // online softmax (stats per logical row v + 8*hi; 16-lane reductions)
    float rmax[8];
#pragma unroll
    for (int v = 0; v < 8; ++v) rmax[v] = -3.0e38f;
#pragma unroll
    for (int nf = 0; nf < 4; ++nf)
#pragma unroll
      for (int v = 0; v < 8; ++v) {
        float s = S[nf][v] * scale;
        S[nf][v] = s;
        rmax[v] = fmaxf(rmax[v], s);
      }
#pragma unroll
    for (int off = 1; off < 16; off <<= 1)
#pragma unroll
      for (int v = 0; v < 8; ++v)
        rmax[v] = fmaxf(rmax[v], __shfl_xor(rmax[v], off, 32));

    float alpha[8], ladd[8];
#pragma unroll
    for (int v = 0; v < 8; ++v) {
      float mn = fmaxf(mrow[v], rmax[v]);
      alpha[v] = __expf(mrow[v] - mn);
      mrow[v]  = mn;
      ladd[v]  = 0.0f;
    }
#pragma unroll
    for (int nf = 0; nf < 4; ++nf)
#pragma unroll
      for (int v = 0; v < 8; ++v) {
        float p = __expf(S[nf][v] - mrow[v]);
        ladd[v] += p;
        Ps[wave][(v + 8 * hi) * 64 + nf * 16 + nlo] = f2bf(p);  // C-layout -> row-major
      }
#pragma unroll
    for (int off = 1; off < 16; off <<= 1)
#pragma unroll
      for (int v = 0; v < 8; ++v)
        ladd[v] += __shfl_xor(ladd[v], off, 32);
#pragma unroll
    for (int v = 0; v < 8; ++v) lsum[v] = lsum[v] * alpha[v] + ladd[v];
#pragma unroll
    for (int nf = 0; nf < 8; ++nf)
#pragma unroll
      for (int v = 0; v < 8; ++v) O[nf][v] *= alpha[v];

    __syncthreads();   // P visible before re-reading as A fragments

    // O(16x128) += P(16x64) @ V(64x128)
#pragma unroll
    for (int kc = 0; kc < 2; ++kc) {
      v16bf ap = load_frag_a(Ps[wave] + kc * 32, 64, lane);
#pragma unroll
      for (int nf = 0; nf < 8; ++nf) {
        v16bf bv = load_frag_b(Vt + (nf * 16) * 64 + kc * 32, 64, lane);
        O[nf] = wmma_bf16(ap, bv, O[nf]);
      }
    }
  }

  // normalize and write attn[1536][3072] (fp32 for the output projections)
#pragma unroll
  for (int v = 0; v < 8; ++v) {
    float inv = 1.0f / lsum[v];
    int row = qt * 64 + wave * 16 + v + 8 * hi;
#pragma unroll
    for (int nf = 0; nf < 8; ++nf)
      attn[(size_t)row * DM + h * HDd + nf * 16 + nlo] = O[nf][v] * inv;
  }
}

// ---------------------------------------------------------------------------
extern "C" void kernel_launch(void* const* d_in, const int* in_sizes, int n_in,
                              void* d_out, int out_size, void* d_ws, size_t ws_size,
                              hipStream_t stream) {
  (void)in_sizes; (void)n_in; (void)out_size; (void)ws_size;
  const float* img        = (const float*)d_in[0];
  const float* txt        = (const float*)d_in[1];
  const float* vec        = (const float*)d_in[2];
  const float* pe         = (const float*)d_in[3];
  const float* mod_img_w  = (const float*)d_in[4];
  const float* mod_img_b  = (const float*)d_in[5];
  const float* mod_txt_w  = (const float*)d_in[6];
  const float* mod_txt_b  = (const float*)d_in[7];
  const float* qkv_img_w  = (const float*)d_in[8];
  const float* qkv_img_b  = (const float*)d_in[9];
  const float* qkv_txt_w  = (const float*)d_in[10];
  const float* qkv_txt_b  = (const float*)d_in[11];
  const float* norm_q_w   = (const float*)d_in[12];
  const float* norm_k_w   = (const float*)d_in[13];
  const float* norm_aq_w  = (const float*)d_in[14];
  const float* norm_ak_w  = (const float*)d_in[15];
  const float* out_img_w  = (const float*)d_in[16];
  const float* out_img_b  = (const float*)d_in[17];
  const float* out_txt_w  = (const float*)d_in[18];
  const float* out_txt_b  = (const float*)d_in[19];
  const float* fc1_img_w  = (const float*)d_in[20];
  const float* fc1_img_b  = (const float*)d_in[21];
  const float* fc2_img_w  = (const float*)d_in[22];
  const float* fc2_img_b  = (const float*)d_in[23];
  const float* fc1_txt_w  = (const float*)d_in[24];
  const float* fc1_txt_b  = (const float*)d_in[25];
  const float* fc2_txt_w  = (const float*)d_in[26];
  const float* fc2_txt_b  = (const float*)d_in[27];

  float* out = (float*)d_out;                 // img [1024x3072] then txt [512x3072]
  float* ws  = (float*)d_ws;

  // workspace layout (float units); ff reuses the qkv region (consumed by qkv_post).
  float* modv  = ws;                            // 36,864
  float* act   = modv + 36864;                  // 3,145,728
  float* qkvi  = act  + (size_t)LIi * DM;       // 9,437,184
  float* qkvt  = qkvi + (size_t)LIi * 3 * DM;   // 4,718,592
  float* attnb = qkvt + (size_t)LTt * 3 * DM;   // 4,718,592
  unsigned short* qbuf = (unsigned short*)(attnb + (size_t)LLl * DM);  // 4,718,592 bf16
  unsigned short* kbuf = qbuf + (size_t)LLl * DM;                      // 4,718,592 bf16
  unsigned short* vbT  = kbuf + (size_t)LLl * DM;                      // 4,718,592 bf16
  float* ff    = qkvi;                          // 12,582,912 (fits qkvi+qkvt = 14.1M)

  // 1) modulation vectors
  mod_gemv_kernel<<<576, 256, 0, stream>>>(vec, mod_img_w, mod_img_b,
                                           mod_txt_w, mod_txt_b, modv);

  // 2) LN+mod, qkv projections  (mod chunks: 0=shift1 1=scale1 2=gate1 3=shift2 4=scale2 5=gate2)
  ln_mod_kernel<<<LIi, 256, 0, stream>>>(img, modv + 0 * DM, modv + 1 * DM, act);
  wmma_gemm<<<dim3(72, 8), 256, 0, stream>>>(act, DM, qkv_img_w, 3 * DM, qkv_img_b,
                                             qkvi, 3 * DM, nullptr, nullptr, DM, 0);
  ln_mod_kernel<<<LTt, 256, 0, stream>>>(txt, modv + 6 * DM + 0 * DM, modv + 6 * DM + 1 * DM, act);
  wmma_gemm<<<dim3(72, 4), 256, 0, stream>>>(act, DM, qkv_txt_w, 3 * DM, qkv_txt_b,
                                             qkvt, 3 * DM, nullptr, nullptr, DM, 0);

  // 3) RMS-norm + RoPE + concat [txt;img] -> bf16 q/k, bf16 transposed v
  qkv_post_kernel<<<dim3(LLl, Hh), 128, 0, stream>>>(qkvi, qkvt, norm_q_w, norm_k_w,
                                                     norm_aq_w, norm_ak_w, pe,
                                                     qbuf, kbuf, vbT);

  // 4) flash attention (async-to-LDS staging)
  flash_attn_kernel<<<dim3(24, Hh), 128, 0, stream>>>(qbuf, kbuf, vbT, attnb);

  // 5) output projections with residual + per-channel gate -> d_out
  float* oimg = out;
  float* otxt = out + (size_t)LIi * DM;
  wmma_gemm<<<dim3(24, 8), 256, 0, stream>>>(attnb + (size_t)LTt * DM, DM, out_img_w, DM,
                                             out_img_b, oimg, DM, img, modv + 2 * DM, DM, 2);
  wmma_gemm<<<dim3(24, 4), 256, 0, stream>>>(attnb, DM, out_txt_w, DM,
                                             out_txt_b, otxt, DM, txt, modv + 6 * DM + 2 * DM, DM, 2);

  // 6) img MLP: LN+mod -> gelu(fc1) -> fc2 residual+gate
  ln_mod_kernel<<<LIi, 256, 0, stream>>>(oimg, modv + 3 * DM, modv + 4 * DM, act);
  wmma_gemm<<<dim3(96, 8), 256, 0, stream>>>(act, DM, fc1_img_w, MLPd, fc1_img_b,
                                             ff, MLPd, nullptr, nullptr, DM, 1);
  wmma_gemm<<<dim3(24, 8), 256, 0, stream>>>(ff, MLPd, fc2_img_w, DM, fc2_img_b,
                                             oimg, DM, oimg, modv + 5 * DM, MLPd, 2);

  // 7) txt MLP
  ln_mod_kernel<<<LTt, 256, 0, stream>>>(otxt, modv + 6 * DM + 3 * DM, modv + 6 * DM + 4 * DM, act);
  wmma_gemm<<<dim3(96, 4), 256, 0, stream>>>(act, DM, fc1_txt_w, MLPd, fc1_txt_b,
                                             ff, MLPd, nullptr, nullptr, DM, 1);
  wmma_gemm<<<dim3(24, 4), 256, 0, stream>>>(ff, MLPd, fc2_txt_w, DM, fc2_txt_b,
                                             otxt, DM, otxt, modv + 6 * DM + 5 * DM, MLPd, 2);
}
